// Model_36163624632611
// MI455X (gfx1250) — compile-verified, matrix operand order checked by reading, not run on hardware
//
#include <hip/hip_runtime.h>
#include <hip/hip_bf16.h>
#include <cstddef>
#include <cstdint>

#define B_SZ 16
#define N_ANC 25200
#define NCLS 80
#define KDET 300
#define PPT 128
#define ST 1120           // padded channel stride of the snake state stack
#define CONF_T 0.2f
#define IOU_T 0.3f
#define MAXBUF 8192       // halfs per weight staging buffer (256 rows x 32)

typedef __attribute__((ext_vector_type(16))) __bf16 v16bf;
typedef __attribute__((ext_vector_type(8)))  float  v8f;
typedef __attribute__((ext_vector_type(4)))  int    v4i;

// ---- CDNA5 async global->LDS path (guarded; falls back to plain LDS stores)
#if defined(__gfx1250__) && defined(__has_builtin)
#if __has_builtin(__builtin_amdgcn_global_load_async_to_lds_b128) && \
    __has_builtin(__builtin_amdgcn_s_wait_asynccnt)
#define USE_ASYNC_LDS 1
#endif
#endif
#if defined(USE_ASYNC_LDS)
typedef __attribute__((address_space(1))) v4i gv4i;   // global int4
typedef __attribute__((address_space(3))) v4i lv4i;   // LDS int4
#endif

__device__ __forceinline__ v8f wmma_bf16(v16bf a, v16bf b, v8f c) {
  return __builtin_amdgcn_wmma_f32_16x16x32_bf16(false, a, false, b, (short)0, c,
                                                 false, false);
}

// Load a 16x32 bf16 operand fragment. Caller passes the pointer to element
// (row_of_this_lane, k=0); K is contiguous. Mapping per CDNA5 ISA 7.12.2:
// lane L: row = L%16, group g = L/16; VGPR v holds K = {base, base+1},
// base = (v<4 ? 0:16) + (v&3)*2 + g*8.
__device__ __forceinline__ v16bf load_frag(const __bf16* p) {
  int g = (threadIdx.x >> 4) & 1;
  v16bf f;
#pragma unroll
  for (int v = 0; v < 8; ++v) {
    int k = ((v & 4) << 2) + ((v & 3) << 1) + (g << 3);
    f[2 * v]     = p[k];
    f[2 * v + 1] = p[k + 1];
  }
  return f;
}

// Cooperative copy of an Nout x 32-half weight block into LDS (row stride K).
__device__ __forceinline__ void copy_block(const __bf16* g, __bf16* l,
                                           int rows, int K) {
  int total = rows * 4;                       // 16B chunks
  for (int c = threadIdx.x; c < total; c += 256) {
    int row = c >> 2, sub = c & 3;
    const __bf16* gp = g + (size_t)row * K + sub * 8;
    __bf16* lp = l + row * 32 + sub * 8;
#if defined(USE_ASYNC_LDS)
    __builtin_amdgcn_global_load_async_to_lds_b128(
        (gv4i*)(uintptr_t)gp, (lv4i*)(uintptr_t)lp, 0, 0);
#else
    *(float4*)lp = *(const float4*)gp;
#endif
  }
}
__device__ __forceinline__ void wait_copies() {
#if defined(USE_ASYNC_LDS)
  __builtin_amdgcn_s_wait_asynccnt(0);
#endif
  __syncthreads();
}

// ---------------------------------------------------------------- NMS path
__global__ void k_score(const float* __restrict__ pred, float* scores, float* cls) {
  int t = blockIdx.x * blockDim.x + threadIdx.x;
  if (t >= B_SZ * N_ANC) return;
  const float* r = pred + (size_t)t * (5 + NCLS);
  float obj = r[4];
  float best = -1.f; int bi = 0;
  for (int c = 0; c < NCLS; ++c) { float v = r[5 + c]; if (v > best) { best = v; bi = c; } }
  float s = obj * best;
  scores[t] = (s > CONF_T) ? s : 0.f;
  cls[t] = (float)bi;
}

__global__ __launch_bounds__(256) void k_topk(float* scores, int* topi, float* tops) {
  int b = blockIdx.x;
  float* s = scores + (size_t)b * N_ANC;
  __shared__ float sv[256];
  __shared__ int   si[256];
  for (int r = 0; r < KDET; ++r) {
    float bv = -1.f; int bi = 0;
    for (int i = threadIdx.x; i < N_ANC; i += 256) {
      float v = s[i];
      if (v > bv) { bv = v; bi = i; }
    }
    sv[threadIdx.x] = bv; si[threadIdx.x] = bi;
    __syncthreads();
    for (int st = 128; st > 0; st >>= 1) {
      if (threadIdx.x < st && sv[threadIdx.x + st] > sv[threadIdx.x]) {
        sv[threadIdx.x] = sv[threadIdx.x + st];
        si[threadIdx.x] = si[threadIdx.x + st];
      }
      __syncthreads();
    }
    if (threadIdx.x == 0) {
      topi[b * KDET + r] = si[0];
      tops[b * KDET + r] = fmaxf(sv[0], 0.f);
      s[si[0]] = -1.f;
    }
    __syncthreads();
  }
}

__global__ void k_boxes(const float* __restrict__ pred, const int* topi,
                        const float* tops, const float* cls,
                        float* boxes, float* clsk, float* outDet, float* outCt) {
  int t = blockIdx.x * blockDim.x + threadIdx.x;
  if (t >= B_SZ * KDET) return;
  int b = t / KDET;
  int i = topi[t];
  const float* r = pred + ((size_t)b * N_ANC + i) * (5 + NCLS);
  float cx = r[0], cy = r[1], w = r[2], h = r[3];
  float x1 = cx - 0.5f * w, y1 = cy - 0.5f * h;
  float x2 = cx + 0.5f * w, y2 = cy + 0.5f * h;
  boxes[t * 4 + 0] = x1; boxes[t * 4 + 1] = y1;
  boxes[t * 4 + 2] = x2; boxes[t * 4 + 3] = y2;
  float cv = cls[(size_t)b * N_ANC + i];
  clsk[t] = cv;
  outDet[t * 6 + 0] = x1; outDet[t * 6 + 1] = y1;
  outDet[t * 6 + 2] = x2; outDet[t * 6 + 3] = y2;
  outDet[t * 6 + 4] = tops[t]; outDet[t * 6 + 5] = cv;
  outCt[t * 2 + 0] = (x1 + x2) * 0.125f;   // center of boxes/4
  outCt[t * 2 + 1] = (y1 + y2) * 0.125f;
}

__global__ __launch_bounds__(256) void k_nms(const float* boxes, const float* tops,
                                             const float* clsk, float* validf,
                                             float* outValid) {
  int b = blockIdx.x;
  __shared__ float bx[KDET][4];
  __shared__ float ar[KDET];
  __shared__ int   keep[KDET];
  for (int i = threadIdx.x; i < KDET; i += blockDim.x) {
    float off = clsk[b * KDET + i] * 4096.f;
    float x1 = boxes[(b * KDET + i) * 4 + 0] + off;
    float y1 = boxes[(b * KDET + i) * 4 + 1] + off;
    float x2 = boxes[(b * KDET + i) * 4 + 2] + off;
    float y2 = boxes[(b * KDET + i) * 4 + 3] + off;
    bx[i][0] = x1; bx[i][1] = y1; bx[i][2] = x2; bx[i][3] = y2;
    ar[i] = (x2 - x1) * (y2 - y1);
    keep[i] = 1;
  }
  __syncthreads();
  for (int i = 0; i < KDET; ++i) {
    if (keep[i]) {
      for (int j = i + 1 + threadIdx.x; j < KDET; j += blockDim.x) {
        float xx1 = fmaxf(bx[i][0], bx[j][0]);
        float yy1 = fmaxf(bx[i][1], bx[j][1]);
        float xx2 = fminf(bx[i][2], bx[j][2]);
        float yy2 = fminf(bx[i][3], bx[j][3]);
        float inter = fmaxf(xx2 - xx1, 0.f) * fmaxf(yy2 - yy1, 0.f);
        float iou = inter / (ar[i] + ar[j] - inter + 1e-9f);
        if (iou > IOU_T) keep[j] = 0;
      }
    }
    __syncthreads();
  }
  for (int i = threadIdx.x; i < KDET; i += blockDim.x) {
    float v = (keep[i] && tops[b * KDET + i] > CONF_T) ? 1.f : 0.f;
    validf[b * KDET + i] = v;
    outValid[b * KDET + i] = v;
  }
}

// --------------------------------------------------- weight packing -> bf16
__global__ void k_pack_head(const float* w, __bf16* o) {      // [9][128][96]
  int t = blockIdx.x * blockDim.x + threadIdx.x;
  if (t >= 9 * 128 * 96) return;
  int j = t / (128 * 96), r = t % (128 * 96), oc = r / 96, c = r % 96;
  float v = (c < 66) ? w[((size_t)oc * 66 + c) * 9 + j] : 0.f;
  o[t] = (__bf16)v;
}
__global__ void k_pack_res(const float* w, __bf16* o) {       // [7][9][128][128]
  int t = blockIdx.x * blockDim.x + threadIdx.x;
  if (t >= 7 * 9 * 128 * 128) return;
  int l = t / (9 * 128 * 128), r = t % (9 * 128 * 128);
  int j = r / (128 * 128); r %= (128 * 128);
  int oc = r / 128, c = r % 128;
  o[t] = (__bf16)w[(((size_t)l * 128 + oc) * 128 + c) * 9 + j];
}
__global__ void k_pack_fuse(const float* w, __bf16* o) {      // [256][1120]
  int t = blockIdx.x * blockDim.x + threadIdx.x;
  if (t >= 256 * ST) return;
  int oc = t / ST, k = t % ST;
  float v = 0.f;
  if (k < 66)      v = w[(size_t)oc * 1090 + k];
  else if (k >= 96) v = w[(size_t)oc * 1090 + 66 + (k - 96)];
  o[t] = (__bf16)v;
}
__global__ void k_pack_p1(const float* w, __bf16* o) {        // [64][256]
  int t = blockIdx.x * blockDim.x + threadIdx.x;
  if (t >= 64 * 256) return;
  o[t] = (__bf16)w[t];
}
__global__ void k_pack_p2(const float* w, __bf16* o) {        // [16][64], rows>=2 zero
  int t = blockIdx.x * blockDim.x + threadIdx.x;
  if (t >= 16 * 64) return;
  int oc = t / 64, c = t % 64;
  o[t] = (__bf16)((oc < 2) ? w[oc * 64 + c] : 0.f);
}
__global__ void k_pack_conv(const float* w, __bf16* o) {      // [9][64][128]
  int t = blockIdx.x * blockDim.x + threadIdx.x;
  if (t >= 9 * 64 * 128) return;
  int j = t / (64 * 128), r = t % (64 * 128), oc = r / 128, c = r % 128;
  int dy = j / 3, dx = j % 3;
  o[t] = (__bf16)w[(((size_t)oc * 128 + c) * 3 + dy) * 3 + dx];
}

// ----------------------------- upsample2x + 3x3 conv (128->64) + SiLU -> cf
__global__ __launch_bounds__(256) void k_upconv(const float* __restrict__ feat,
                                                const __bf16* __restrict__ pw,
                                                const float* __restrict__ bias,
                                                __bf16* __restrict__ cf) {
  int ox0 = blockIdx.x * 16, oy0 = blockIdx.y * 4, b = blockIdx.z;
  __shared__ __bf16 up[6 * 18 * 128];                 // halo patch, bf16
  const float* fb = feat + (size_t)b * 128 * 80 * 80;
  for (int idx = threadIdx.x; idx < 6 * 18 * 128; idx += 256) {
    int pos = idx >> 7, ch = idx & 127;
    int pr = pos / 18, pc = pos % 18;
    int uy = oy0 - 1 + pr; uy = uy < 0 ? 0 : (uy > 159 ? 159 : uy);
    int ux = ox0 - 1 + pc; ux = ux < 0 ? 0 : (ux > 159 ? 159 : ux);
    float sy = fminf(fmaxf(uy * 0.5f - 0.25f, 0.f), 79.f);
    float sx = fminf(fmaxf(ux * 0.5f - 0.25f, 0.f), 79.f);
    int y0 = (int)sy; y0 = y0 > 78 ? 78 : y0; float fy = sy - y0;
    int x0 = (int)sx; x0 = x0 > 78 ? 78 : x0; float fx = sx - x0;
    const float* fc = fb + (size_t)ch * 6400;
    float v00 = fc[y0 * 80 + x0],       v01 = fc[y0 * 80 + x0 + 1];
    float v10 = fc[(y0 + 1) * 80 + x0], v11 = fc[(y0 + 1) * 80 + x0 + 1];
    float v = v00 * (1 - fx) * (1 - fy) + v01 * fx * (1 - fy) +
              v10 * (1 - fx) * fy + v11 * fx * fy;
    up[pos * 128 + ch] = (__bf16)v;
  }
  __syncthreads();
  int lane = threadIdx.x & 31, wv = threadIdx.x >> 5;
  int g = (lane >> 4) & 1, ln = lane & 15;
  v8f acc[2];
  for (int s = 0; s < 2; ++s)
    for (int v = 0; v < 8; ++v) acc[s][v] = 0.f;
  for (int j = 0; j < 9; ++j) {
    int dy = j / 3, dx = j % 3;
    for (int ks = 0; ks < 4; ++ks) {
      for (int s = 0; s < 2; ++s) {
        int tile = wv * 2 + s, mt = tile >> 2, nt = tile & 3;
        v16bf a  = load_frag(up + ((mt + dy) * 18 + (ln + dx)) * 128 + ks * 32);
        v16bf wb = load_frag(pw + ((size_t)(j * 64 + nt * 16 + ln)) * 128 + ks * 32);
        acc[s] = wmma_bf16(a, wb, acc[s]);
      }
    }
  }
  for (int s = 0; s < 2; ++s) {
    int tile = wv * 2 + s, mt = tile >> 2, nt = tile & 3;
    int oc = nt * 16 + ln;
    float bv = bias[oc];
    int oy = oy0 + mt;
    for (int v = 0; v < 8; ++v) {
      int ox = ox0 + v + 8 * g;
      float x = acc[s][v] + bv;
      float sv = 1.f / (1.f + __expf(-x));
      cf[(((size_t)b * 64 + oc) * 160 + oy) * 160 + ox] = (__bf16)(x * sv);
    }
  }
}

// ----------------------------------------------------------- snake layers
// One circular-conv (or 1x1) layer as WMMA GEMM over the LDS state stack.
// Weights are staged through LDS (double-buffered, async-to-LDS when
// available) so the 8 waves of the WG share one copy instead of 8 re-reads.
__device__ __forceinline__ void snake_layer(__bf16* stk, __bf16* wbuf,
                                            const __bf16* pw, const float* bias,
                                            int co_in, int K, int co_out, int Nout,
                                            int taps, int dil, bool residual,
                                            bool preBarrier, int lane, int wv) {
  int g = (lane >> 4) & 1, ln = lane & 15;
  int NT = Nout >> 4;
  int mt = wv;                               // 8 waves == 8 M-tiles (128 points)
  int KS = K >> 5;
  int S = taps * KS;
  v8f acc[16];
  for (int nt = 0; nt < NT; ++nt)
    for (int v = 0; v < 8; ++v) acc[nt][v] = 0.f;
  copy_block(pw, wbuf, Nout, K);             // preload block 0 (t=0, ks=0)
  wait_copies();
  for (int s = 0; s < S; ++s) {
    int t = s / KS, ks = s % KS;
    if (s + 1 < S) {
      int t2 = (s + 1) / KS, ks2 = (s + 1) % KS;
      copy_block(pw + (size_t)t2 * Nout * K + ks2 * 32,
                 wbuf + ((s + 1) & 1) * MAXBUF, Nout, K);
    }
    const __bf16* bb = wbuf + (s & 1) * MAXBUF;
    int shift = dil * (t - (taps >> 1));     // taps=9 -> t-4 ; taps=1 -> 0
    int prow = (mt * 16 + ln + shift) & 127;
    v16bf a = load_frag(stk + prow * ST + co_in + ks * 32);
    for (int nt = 0; nt < NT; ++nt) {
      v16bf wfr = load_frag(bb + (nt * 16 + ln) * 32);
      acc[nt] = wmma_bf16(a, wfr, acc[nt]);
    }
    wait_copies();                           // copy(s+1) landed; all waves done s
  }
  if (preBarrier) __syncthreads();           // fuse overwrites a region it reads
  for (int nt = 0; nt < NT; ++nt) {
    int c = nt * 16 + ln;
    float bv = bias[c];
    for (int v = 0; v < 8; ++v) {
      int p = mt * 16 + v + 8 * g;
      float x = fmaxf(acc[nt][v] + bv, 0.f); // relu on every layer we route here
      if (residual) x += (float)stk[p * ST + co_in + c];
      stk[p * ST + co_out + c] = (__bf16)x;
    }
  }
  __syncthreads();
}

__global__ __launch_bounds__(256) void k_snake(
    const __bf16* __restrict__ cf, const float* __restrict__ boxes,
    const float* __restrict__ validf,
    const __bf16* pwh, const __bf16* pwr, const __bf16* pwf,
    const __bf16* pw1, const __bf16* pw2,
    const float* bh, const float* br, const float* bfu,
    const float* b1, const float* b2, float* outPts) {
  extern __shared__ char smem[];
  __bf16* stk  = (__bf16*)smem;                         // [128][1120] bf16
  float*  lpts = (float*)(smem + (size_t)128 * ST * 2); // [128][2]
  float*  misc = lpts + 256;                            // 16 floats
  __bf16* wbuf = (__bf16*)(smem + (size_t)128 * ST * 2 + 1088); // 2*MAXBUF halfs
  const int DILA[7] = {1, 1, 1, 2, 2, 4, 4};
  int inst = blockIdx.x, b = inst / KDET, tid = threadIdx.x;
  int lane = tid & 31, wv = tid >> 5;
  if (tid == 0) {
    float x1 = boxes[inst * 4 + 0] * 0.25f, y1 = boxes[inst * 4 + 1] * 0.25f;
    float x2 = boxes[inst * 4 + 2] * 0.25f, y2 = boxes[inst * 4 + 3] * 0.25f;
    misc[0] = (x1 + x2) * 0.5f; misc[1] = (y1 + y2) * 0.5f;
    misc[2] = fmaxf((x2 - x1) * 0.5f, 1e-3f);
    misc[3] = fmaxf((y2 - y1) * 0.5f, 1e-3f);
  }
  __syncthreads();
  float ctx = misc[0], cty = misc[1], hx = misc[2], hy = misc[3];
  if (tid < 128) {
    float th = tid * 0.04908738521234052f;              // 2*pi/128
    lpts[2 * tid]     = ctx + hx * __cosf(th);
    lpts[2 * tid + 1] = cty + hy * __sinf(th);
  }
  const __bf16* cfb = cf + (size_t)b * 64 * 160 * 160;
  for (int it = 0; it < 2; ++it) {
    __syncthreads();
    { // stage z = [sampled 64ch | rel 2ch | zero pad to 96] into stack ch 0..95
      int p = tid & 127, cg = tid >> 7;
      float px = lpts[2 * p], py = lpts[2 * p + 1];
      float xx = fminf(fmaxf(px, 0.f), 159.f);
      float yy = fminf(fmaxf(py, 0.f), 159.f);
      int x0 = (int)xx; x0 = x0 > 158 ? 158 : x0;
      int y0 = (int)yy; y0 = y0 > 158 ? 158 : y0;
      float dx = xx - x0, dy = yy - y0;
      float w00 = (1 - dx) * (1 - dy), w01 = dx * (1 - dy);
      float w10 = (1 - dx) * dy,       w11 = dx * dy;
      for (int c = cg * 48; c < cg * 48 + 48; ++c) {
        float v;
        if (c < 64) {
          const __bf16* f = cfb + (size_t)c * 25600 + y0 * 160 + x0;
          v = (float)f[0] * w00 + (float)f[1] * w01 +
              (float)f[160] * w10 + (float)f[161] * w11;
        } else if (c == 64) v = (px - ctx) / hx;
        else if (c == 65)   v = (py - cty) / hy;
        else v = 0.f;
        stk[p * ST + c] = (__bf16)v;
      }
    }
    __syncthreads();
    snake_layer(stk, wbuf, pwh, bh, 0, 96, 96, 128, 9, 1, false, false, lane, wv);
    for (int l = 0; l < 7; ++l)
      snake_layer(stk, wbuf, pwr + (size_t)l * 9 * 128 * 128, br + l * 128,
                  96 + l * 128, 128, 96 + (l + 1) * 128, 128, 9, DILA[l],
                  true, false, lane, wv);
    snake_layer(stk, wbuf, pwf, bfu, 0, ST, 0, 256, 1, 1, false, true, lane, wv);
    snake_layer(stk, wbuf, pw1, b1, 0, 256, 256, 64, 1, 1, false, false, lane, wv);
    { // p2: 64 -> 2 (padded to 16 cols); update points
      int g = (lane >> 4) & 1, ln = lane & 15, mt = wv;
      v8f acc;
      for (int v = 0; v < 8; ++v) acc[v] = 0.f;
      for (int ks = 0; ks < 2; ++ks) {
        v16bf a  = load_frag(stk + (mt * 16 + ln) * ST + 256 + ks * 32);
        v16bf wb = load_frag(pw2 + ln * 64 + ks * 32);
        acc = wmma_bf16(a, wb, acc);
      }
      __syncthreads();
      if (ln < 2) {
        float bv = b2[ln];
        for (int v = 0; v < 8; ++v) {
          int p = mt * 16 + v + 8 * g;
          lpts[2 * p + ln] += acc[v] + bv;
        }
      }
    }
  }
  __syncthreads();
  float vf = validf[inst];
  if (tid < 128) {
    outPts[((size_t)inst * 128 + tid) * 2]     = lpts[2 * tid] * vf;
    outPts[((size_t)inst * 128 + tid) * 2 + 1] = lpts[2 * tid + 1] * vf;
  }
}

// ------------------------------------------------------------------ launch
extern "C" void kernel_launch(void* const* d_in, const int* in_sizes, int n_in,
                              void* d_out, int out_size, void* d_ws, size_t ws_size,
                              hipStream_t stream) {
  const float* pred   = (const float*)d_in[0];
  const float* feat   = (const float*)d_in[1];
  const float* conv_w = (const float*)d_in[2];
  const float* conv_b = (const float*)d_in[3];
  const float* w_head = (const float*)d_in[4];
  const float* b_head = (const float*)d_in[5];
  const float* w_res  = (const float*)d_in[6];
  const float* b_res  = (const float*)d_in[7];
  const float* w_fuse = (const float*)d_in[8];
  const float* b_fuse = (const float*)d_in[9];
  const float* w_p1   = (const float*)d_in[10];
  const float* b_p1   = (const float*)d_in[11];
  const float* w_p2   = (const float*)d_in[12];
  const float* b_p2   = (const float*)d_in[13];
  float* out = (float*)d_out;

  char* w = (char*)d_ws;
  size_t off = 0;
  auto alloc = [&](size_t bytes) { char* p = w + off; off = (off + bytes + 255) & ~(size_t)255; return p; };
  float* ws_scores = (float*)alloc((size_t)B_SZ * N_ANC * 4);
  float* ws_cls    = (float*)alloc((size_t)B_SZ * N_ANC * 4);
  int*   ws_topi   = (int*)  alloc((size_t)B_SZ * KDET * 4);
  float* ws_tops   = (float*)alloc((size_t)B_SZ * KDET * 4);
  float* ws_boxes  = (float*)alloc((size_t)B_SZ * KDET * 16);
  float* ws_clsk   = (float*)alloc((size_t)B_SZ * KDET * 4);
  float* ws_valid  = (float*)alloc((size_t)B_SZ * KDET * 4);
  __bf16* pw_head  = (__bf16*)alloc((size_t)9 * 128 * 96 * 2);
  __bf16* pw_res   = (__bf16*)alloc((size_t)7 * 9 * 128 * 128 * 2);
  __bf16* pw_fuse  = (__bf16*)alloc((size_t)256 * ST * 2);
  __bf16* pw_p1    = (__bf16*)alloc((size_t)64 * 256 * 2);
  __bf16* pw_p2    = (__bf16*)alloc((size_t)16 * 64 * 2);
  __bf16* pw_conv  = (__bf16*)alloc((size_t)9 * 64 * 128 * 2);
  __bf16* cf       = (__bf16*)alloc((size_t)B_SZ * 64 * 160 * 160 * 2);
  (void)in_sizes; (void)n_in; (void)out_size; (void)ws_size;

  float* outDet   = out;                        // [16,300,6]
  float* outCt    = out + 28800;                // [16,300,2]
  float* outPts   = out + 38400;                // [16,300,128,2]
  float* outValid = out + 1267200;              // [16,300]

  k_score<<<(B_SZ * N_ANC + 255) / 256, 256, 0, stream>>>(pred, ws_scores, ws_cls);
  k_topk <<<B_SZ, 256, 0, stream>>>(ws_scores, ws_topi, ws_tops);
  k_boxes<<<(B_SZ * KDET + 255) / 256, 256, 0, stream>>>(pred, ws_topi, ws_tops,
                                                         ws_cls, ws_boxes, ws_clsk,
                                                         outDet, outCt);
  k_nms  <<<B_SZ, 256, 0, stream>>>(ws_boxes, ws_tops, ws_clsk, ws_valid, outValid);

  k_pack_head<<<(9 * 128 * 96 + 255) / 256, 256, 0, stream>>>(w_head, pw_head);
  k_pack_res <<<(7 * 9 * 128 * 128 + 255) / 256, 256, 0, stream>>>(w_res, pw_res);
  k_pack_fuse<<<(256 * ST + 255) / 256, 256, 0, stream>>>(w_fuse, pw_fuse);
  k_pack_p1  <<<(64 * 256 + 255) / 256, 256, 0, stream>>>(w_p1, pw_p1);
  k_pack_p2  <<<(16 * 64 + 255) / 256, 256, 0, stream>>>(w_p2, pw_p2);
  k_pack_conv<<<(9 * 64 * 128 + 255) / 256, 256, 0, stream>>>(conv_w, pw_conv);

  k_upconv<<<dim3(10, 40, B_SZ), 256, 0, stream>>>(feat, pw_conv, conv_b, cf);

  // LDS: state stack 286,720 + pts/misc 1,088 + 2x16KB weight buffers = 320,576 B
  size_t snake_lds = (size_t)128 * ST * 2 + 1088 + 2 * (size_t)MAXBUF * 2;
  k_snake<<<B_SZ * KDET, 256, snake_lds, stream>>>(cf, ws_boxes, ws_valid,
                                                   pw_head, pw_res, pw_fuse, pw_p1,
                                                   pw_p2, b_head, b_res, b_fuse,
                                                   b_p1, b_p2, outPts);
}
